// HydraScaleLM_46428596470336
// MI455X (gfx1250) — matrix operand order repaired and checked by reading, not compiled, not run on hardware
//
#include <hip/hip_runtime.h>
#include <hip/hip_bf16.h>
#include <math.h>

// ---------------- model dims ----------------
constexpr int kE = 768;          // embed dim
constexpr int kN = 16;           // ssm state dim
constexpr int kK = 4;            // d_conv
constexpr int kR = 48;           // dt_rank
constexpr int kDepth = 4;
constexpr int kV = 32000;        // vocab
constexpr int kB = 2;
constexpr int kL = 1024;
constexpr int kH = 4 * kE;       // 3072
constexpr int kP = kR + 2 * kN;  // 80 (params dim, unpadded)
constexpr int kPp = 128;         // params dim padded to N-tile multiple
constexpr int kRp = 64;          // dt_rank padded to K-tile multiple
constexpr int kM = kB * kL;      // 2048 rows

typedef __attribute__((ext_vector_type(16))) _Float16 v16h;
typedef __attribute__((ext_vector_type(8)))  float    v8f;

__device__ __forceinline__ float siluf(float x) { return x * (1.0f / (1.0f + expf(-x))); }
__device__ __forceinline__ float softplusf(float x) { return (x > 20.0f) ? x : log1pf(expf(x)); }
__device__ __forceinline__ float geluf(float x) { return 0.5f * x * (1.0f + erff(x * 0.70710678118654752f)); }

// gfx1250 async global->LDS copy (16B per lane), tracked by ASYNCcnt.
__device__ __forceinline__ void async_copy_b128(unsigned lds_addr, const void* gaddr) {
    asm volatile("global_load_async_to_lds_b128 %0, %1, off"
                 :: "v"(lds_addr), "v"(gaddr) : "memory");
}
__device__ __forceinline__ void wait_async0() {
    asm volatile("s_wait_asynccnt 0x0" ::: "memory");
}
__device__ __forceinline__ unsigned lds_off(const void* p) {
    return (unsigned)(unsigned long long)p;  // addrspace(3) pointer -> 32-bit LDS offset
}

// ---------------- weight convert + zero-pad:  dst[n*Kp+k] = (n<N && k<K) ? f16(src[n*K+k]) : 0
__global__ __launch_bounds__(256)
void cvt_pad_kernel(const float* __restrict__ src, _Float16* __restrict__ dst,
                    int Nsrc, int Ksrc, int Kp, long total)
{
    long idx = (long)blockIdx.x * 256 + threadIdx.x;
    if (idx >= total) return;
    int  k = (int)(idx % Kp);
    long n = idx / Kp;
    float v = (n < Nsrc && k < Ksrc) ? src[n * (long)Ksrc + k] : 0.0f;
    dst[idx] = (_Float16)v;
}

// ---------------- WMMA GEMM: C[M,N] = A[M,K] @ W[N,K]^T (+epilogue) ----------------
// Requirements (guaranteed by caller): M%128==0, N%64==0, K%32==0, A/W are f16,
// all pointers 16B aligned, lda/ldw multiples of 8.
enum { EPI_NONE = 0, EPI_BIAS = 1, EPI_SOFTPLUS = 2, EPI_GELU = 3, EPI_BIAS_RES = 4, EPI_RES = 5 };

template <int EPI>
__global__ __launch_bounds__(256)
void gemm_wmma(const _Float16* __restrict__ A, int lda,
               const _Float16* __restrict__ W, int ldw,
               float* __restrict__ C, _Float16* __restrict__ C16, int ldc,
               int K,
               const float* __restrict__ bias,
               const float* __restrict__ res, int ldres)
{
    // block tile: 128 (M) x 64 (N), K-step 32; 8 waves, wave w owns rows [w*16, w*16+16)
    // double-buffered LDS, filled by async global->LDS loads (no VGPR staging)
    __shared__ __align__(16) _Float16 As[2][128 * 32];  // 2 x 8 KB
    __shared__ __align__(16) _Float16 Ws[2][64 * 32];   // 2 x 4 KB

    const int tid  = threadIdx.x;
    const int wv   = tid >> 5;         // wave id 0..7
    const int lane = tid & 31;
    const int l15  = lane & 15;
    const int kh   = lane >> 4;        // lane-half selector
    const int m0   = blockIdx.y * 128;
    const int n0   = blockIdx.x * 64;

    // staging slices: A -> thread t covers row t/2, 16 halfs at col (t&1)*16 (2 x b128)
    //                 W -> thread t covers row t/4,  8 halfs at col (t&3)*8  (1 x b128)
    const int ar = tid >> 1, ac = (tid & 1) * 16;
    const int wr = tid >> 2, wc = (tid & 3) * 8;
    const _Float16* aptr = A + (size_t)(m0 + ar) * lda + ac;
    const _Float16* wptr = W + (size_t)(n0 + wr) * ldw + wc;

    v8f acc[4] = {};
    const int ktiles = K >> 5;

    // issue tile 0 into buffer 0
    async_copy_b128(lds_off(&As[0][ar * 32 + ac]),     aptr);
    async_copy_b128(lds_off(&As[0][ar * 32 + ac + 8]), aptr + 8);
    async_copy_b128(lds_off(&Ws[0][wr * 32 + wc]),     wptr);

    for (int kt = 0; kt < ktiles; ++kt) {
        const int cur = kt & 1;
        wait_async0();        // our async writes to buffer `cur` have landed in LDS
        __syncthreads();      // ... and so have everyone else's

        if (kt + 1 < ktiles) {
            const int nxt = cur ^ 1;
            const _Float16* ap = aptr + (size_t)(kt + 1) * 32;
            const _Float16* wp = wptr + (size_t)(kt + 1) * 32;
            async_copy_b128(lds_off(&As[nxt][ar * 32 + ac]),     ap);
            async_copy_b128(lds_off(&As[nxt][ar * 32 + ac + 8]), ap + 8);
            async_copy_b128(lds_off(&Ws[nxt][wr * 32 + wc]),     wp);
        }

        // A fragment: 16x32, lanes 0-15 hold K{0..7,16..23}, lanes 16-31 hold K{8..15,24..31}
        v16h af;
        #pragma unroll
        for (int j = 0; j < 16; ++j) {
            int k = (j < 8 ? j : j + 8) + kh * 8;
            af[j] = As[cur][(wv * 16 + l15) * 32 + k];
        }
        // B fragments: 32x16, lane = column N, lanes 0-15 K{0..7,16..23}, lanes 16-31 K{8..15,24..31}
        // load ALL fragments into distinct registers first so the ds_loads pipeline,
        // then issue the four WMMAs back-to-back.
        v16h bf[4];
        #pragma unroll
        for (int bn = 0; bn < 4; ++bn) {
            #pragma unroll
            for (int j = 0; j < 16; ++j) {
                bf[bn][j] = Ws[cur][(bn * 16 + l15) * 32 + ((j < 8 ? j : j + 8) + kh * 8)];
            }
        }
        #pragma unroll
        for (int bn = 0; bn < 4; ++bn) {
            acc[bn] = __builtin_amdgcn_wmma_f32_16x16x32_f16(
                false, af, false, bf[bn], (short)0, acc[bn], false, false);
        }
        // no trailing barrier: next iteration's async writes target the other buffer,
        // and the barrier at its top orders them against this iteration's LDS reads
    }

    // store: C/D layout -> VGPR r: lanes0-15 M=r, lanes16-31 M=r+8; N = lane&15
    #pragma unroll
    for (int bn = 0; bn < 4; ++bn) {
        int n = n0 + bn * 16 + l15;
        #pragma unroll
        for (int r = 0; r < 8; ++r) {
            int m = m0 + wv * 16 + r + kh * 8;
            float v = acc[bn][r];
            if (EPI == EPI_BIAS || EPI == EPI_SOFTPLUS || EPI == EPI_GELU || EPI == EPI_BIAS_RES)
                v += bias[n];
            if (EPI == EPI_SOFTPLUS) v = softplusf(v);
            if (EPI == EPI_GELU)     v = geluf(v);
            if (EPI == EPI_BIAS_RES || EPI == EPI_RES)
                v += res[(size_t)m * ldres + n];
            if (C)   C[(size_t)m * ldc + n]   = v;
            if (C16) C16[(size_t)m * ldc + n] = (_Float16)v;
        }
    }
}

// ---------------- time embedding ----------------
__global__ __launch_bounds__(256)
void temb1_kernel(const int* __restrict__ timesteps, const float* __restrict__ w1,
                  const float* __restrict__ b1, float* __restrict__ th)
{
    __shared__ float emb[kE];
    int b = blockIdx.x;
    float t = (float)timesteps[b];
    for (int e = threadIdx.x; e < kE / 2; e += 256) {
        float f = expf(-9.210340371976184f * (float)e / (float)(kE / 2)); // ln(10000)
        float a = t * f;
        emb[e] = sinf(a);
        emb[e + kE / 2] = cosf(a);
    }
    __syncthreads();
    for (int o = threadIdx.x; o < kH; o += 256) {
        float s = b1[o];
        const float* wr = w1 + (size_t)o * kE;
        for (int e = 0; e < kE; ++e) s += emb[e] * wr[e];
        th[b * kH + o] = siluf(s);
    }
}

__global__ __launch_bounds__(256)
void temb2_kernel(const float* __restrict__ th, const float* __restrict__ w2,
                  const float* __restrict__ b2, float* __restrict__ temb)
{
    __shared__ float hb[kH]; // 12 KB
    int b = blockIdx.x;
    for (int i = threadIdx.x; i < kH; i += 256) hb[i] = th[b * kH + i];
    __syncthreads();
    for (int e = threadIdx.x; e < kE; e += 256) {
        float s = b2[e];
        const float* wr = w2 + (size_t)e * kH;
        for (int k = 0; k < kH; ++k) s += hb[k] * wr[k];
        temb[b * kE + e] = s;
    }
}

// ---------------- embedding gather + broadcast add ----------------
__global__ void embed_kernel(const int* __restrict__ tokens, const float* __restrict__ tok_emb,
                             const float* __restrict__ temb, float* __restrict__ x)
{
    int idx = blockIdx.x * blockDim.x + threadIdx.x;
    if (idx >= kM * kE) return;
    int e = idx % kE;
    int bl = idx / kE;
    int b = bl / kL;
    x[idx] = tok_emb[(size_t)tokens[bl] * kE + e] + temb[b * kE + e];
}

// ---------------- LayerNorm (one block per row), writes f32 + f16 copies ----------------
__global__ __launch_bounds__(256)
void layernorm_kernel(const float* __restrict__ x, const float* __restrict__ g,
                      const float* __restrict__ bta, float* __restrict__ o,
                      _Float16* __restrict__ o16)
{
    __shared__ float red[256];
    int row = blockIdx.x;
    const float* xr = x + (size_t)row * kE;
    float vals[3];
    float s = 0.0f;
    #pragma unroll
    for (int i = 0; i < 3; ++i) { vals[i] = xr[threadIdx.x + i * 256]; s += vals[i]; }
    red[threadIdx.x] = s;
    __syncthreads();
    for (int st = 128; st > 0; st >>= 1) {
        if (threadIdx.x < st) red[threadIdx.x] += red[threadIdx.x + st];
        __syncthreads();
    }
    float mu = red[0] * (1.0f / kE);
    __syncthreads();
    float q = 0.0f;
    #pragma unroll
    for (int i = 0; i < 3; ++i) { float d = vals[i] - mu; q += d * d; }
    red[threadIdx.x] = q;
    __syncthreads();
    for (int st = 128; st > 0; st >>= 1) {
        if (threadIdx.x < st) red[threadIdx.x] += red[threadIdx.x + st];
        __syncthreads();
    }
    float rstd = rsqrtf(red[0] * (1.0f / kE) + 1e-5f);
    #pragma unroll
    for (int i = 0; i < 3; ++i) {
        int e = threadIdx.x + i * 256;
        float v = (vals[i] - mu) * rstd * g[e] + bta[e];
        o[(size_t)row * kE + e] = v;
        o16[(size_t)row * kE + e] = (_Float16)v;
    }
}

// ---------------- depthwise causal conv (K=4) + SiLU ----------------
__global__ void conv_silu_kernel(const float* __restrict__ xn, const float* __restrict__ cw,
                                 float* __restrict__ u)
{
    int idx = blockIdx.x * blockDim.x + threadIdx.x;
    if (idx >= kM * kE) return;
    int e = idx % kE;
    int bl = idx / kE;
    int l = bl % kL;
    int b = bl / kL;
    float s = 0.0f;
    #pragma unroll
    for (int k = 0; k < kK; ++k) {
        int ll = l + k - (kK - 1);
        if (ll >= 0) s += xn[((size_t)(b * kL + ll)) * kE + e] * cw[e * kK + k];
    }
    u[idx] = siluf(s);
}

// ---------------- SSM sequential scan (+ u*D, * silu(z)); writes f16 for out-proj GEMM ----
// replicates reference: h[0]=B[0]*u[0]; h[l]=A[l-1]*h[l-1]+B[l-1]*u[l-1]; y[l]=Cp[l].h[l]
__global__ __launch_bounds__(256)
void ssm_scan_kernel(const float* __restrict__ params /* stride kPp */,
                     const float* __restrict__ dt,
                     const float* __restrict__ u, const float* __restrict__ z,
                     const float* __restrict__ A_log, const float* __restrict__ Dp,
                     _Float16* __restrict__ yg16)
{
    __shared__ float CpS[64][kN];  // 4 KB chunk of C projections (shared across e)
    int b = blockIdx.y;
    int e = blockIdx.x * 256 + threadIdx.x;

    float Avec[kN], h[kN], ap[kN], bp[kN];
    #pragma unroll
    for (int n = 0; n < kN; ++n) {
        Avec[n] = -expf(A_log[e * kN + n]);
        h[n] = 0.0f; ap[n] = 0.0f; bp[n] = 0.0f;
    }
    float De = Dp[e];
    float uprev = 0.0f;

    for (int l = 0; l < kL; ++l) {
        if ((l & 63) == 0) {
            __syncthreads();
            for (int i = threadIdx.x; i < 64 * kN; i += 256) {
                int r = i / kN, n = i % kN;
                CpS[r][n] = params[((size_t)(b * kL + l + r)) * kPp + (kR + kN) + n];
            }
            __syncthreads();
        }
        size_t off = ((size_t)(b * kL + l)) * kE + e;
        float dtv = dt[off];
        float uv  = u[off];
        float zv  = z[off];
        float y = 0.0f;
        #pragma unroll
        for (int n = 0; n < kN; ++n) {
            float a  = Avec[n];
            float ex = expf(dtv * a);
            float bt = (fabsf(a) < 1e-5f) ? dtv : (ex - 1.0f) / (a + 1e-10f);
            float hn = (l == 0) ? bt * uv : (ap[n] * h[n] + bp[n] * uprev);
            h[n] = hn;
            y += CpS[l & 63][n] * hn;
            ap[n] = ex;
            bp[n] = bt;
        }
        uprev = uv;
        y += uv * De;
        yg16[off] = (_Float16)((y) * siluf(zv));
    }
}

// ---------------- host orchestration ----------------
static inline int cdiv(long a, long b) { return (int)((a + b - 1) / b); }

extern "C" void kernel_launch(void* const* d_in, const int* in_sizes, int n_in,
                              void* d_out, int out_size, void* d_ws, size_t ws_size,
                              hipStream_t stream)
{
    const int*   tokens    = (const int*)d_in[0];
    const int*   timesteps = (const int*)d_in[1];
    const float* tok_emb   = (const float*)d_in[2];
    const float* time_w1   = (const float*)d_in[3];
    const float* time_b1   = (const float*)d_in[4];
    const float* time_w2   = (const float*)d_in[5];
    const float* time_b2   = (const float*)d_in[6];
    const float* ln1_g     = (const float*)d_in[7];
    const float* ln1_b     = (const float*)d_in[8];
    const float* z_w       = (const float*)d_in[9];
    const float* p_w       = (const float*)d_in[10];
    const float* conv_w    = (const float*)d_in[11];
    const float* dtp_w     = (const float*)d_in[12];
    const float* dtp_b     = (const float*)d_in[13];
    const float* A_log     = (const float*)d_in[14];
    const float* D_param   = (const float*)d_in[15];
    const float* out_w     = (const float*)d_in[16];
    const float* ln2_g     = (const float*)d_in[17];
    const float* ln2_b     = (const float*)d_in[18];
    const float* mlp_w1    = (const float*)d_in[19];
    const float* mlp_b1    = (const float*)d_in[20];
    const float* mlp_w2    = (const float*)d_in[21];
    const float* mlp_b2    = (const float*)d_in[22];
    const float* lnout_g   = (const float*)d_in[23];
    const float* lnout_b   = (const float*)d_in[24];
    const float* head_w    = (const float*)d_in[25];
    const float* head_b    = (const float*)d_in[26];

    // ---- carve f32 scratch ----
    float* wsf = (float*)d_ws;
    float* x    = wsf; wsf += (size_t)kM * kE;
    float* xn   = wsf; wsf += (size_t)kM * kE;
    float* zb   = wsf; wsf += (size_t)kM * kE;
    float* pb   = wsf; wsf += (size_t)kM * kPp;   // padded params, f32 (scan reads Cp)
    float* dtb  = wsf; wsf += (size_t)kM * kE;
    float* ub   = wsf; wsf += (size_t)kM * kE;
    float* th   = wsf; wsf += (size_t)kB * kH;
    float* temb = wsf; wsf += (size_t)kB * kE;

    // ---- carve f16 scratch (activations) ----
    _Float16* wsh = (_Float16*)wsf;
    _Float16* xn16   = wsh; wsh += (size_t)kM * kE;
    _Float16* pb16   = wsh; wsh += (size_t)kM * kPp;
    _Float16* yg16   = wsh; wsh += (size_t)kM * kE;
    _Float16* hbuf16 = wsh; wsh += (size_t)kM * kH;

    // ---- carve f16 weights (converted + padded each launch; deterministic) ----
    _Float16* zw16  = wsh; wsh += (size_t)kDepth * kE * kE;
    _Float16* pw16  = wsh; wsh += (size_t)kDepth * kPp * kE;
    _Float16* dtw16 = wsh; wsh += (size_t)kDepth * kE * kRp;
    _Float16* ow16  = wsh; wsh += (size_t)kDepth * kE * kE;
    _Float16* w116  = wsh; wsh += (size_t)kDepth * kH * kE;
    _Float16* w216  = wsh; wsh += (size_t)kDepth * kE * kH;
    _Float16* hw16  = wsh; wsh += (size_t)kV * kE;

    // ---- weight conversion (f32 -> f16, zero-padded) ----
    {
        long t;
        t = (long)kDepth * kE * kE;
        cvt_pad_kernel<<<cdiv(t, 256), 256, 0, stream>>>(z_w,   zw16, kDepth * kE, kE, kE, t);
        cvt_pad_kernel<<<cdiv(t, 256), 256, 0, stream>>>(out_w, ow16, kDepth * kE, kE, kE, t);
        t = (long)kDepth * kH * kE;
        cvt_pad_kernel<<<cdiv(t, 256), 256, 0, stream>>>(mlp_w1, w116, kDepth * kH, kE, kE, t);
        t = (long)kDepth * kE * kH;
        cvt_pad_kernel<<<cdiv(t, 256), 256, 0, stream>>>(mlp_w2, w216, kDepth * kE, kH, kH, t);
        t = (long)kV * kE;
        cvt_pad_kernel<<<cdiv(t, 256), 256, 0, stream>>>(head_w, hw16, kV, kE, kE, t);
        for (int i = 0; i < kDepth; ++i) {
            t = (long)kPp * kE;   // rows 80..127 zero
            cvt_pad_kernel<<<cdiv(t, 256), 256, 0, stream>>>(
                p_w + (size_t)i * kP * kE, pw16 + (size_t)i * kPp * kE, kP, kE, kE, t);
            t = (long)kE * kRp;   // cols 48..63 zero
            cvt_pad_kernel<<<cdiv(t, 256), 256, 0, stream>>>(
                dtp_w + (size_t)i * kE * kR, dtw16 + (size_t)i * kE * kRp, kE, kR, kRp, t);
        }
    }

    // ---- embeddings ----
    temb1_kernel<<<kB, 256, 0, stream>>>(timesteps, time_w1, time_b1, th);
    temb2_kernel<<<kB, 256, 0, stream>>>(th, time_w2, time_b2, temb);
    int tot = kM * kE;
    embed_kernel<<<cdiv(tot, 256), 256, 0, stream>>>(tokens, tok_emb, temb, x);

    dim3 gz(kE / 64,  kM / 128);   // (12, 16)
    dim3 gp(kPp / 64, kM / 128);   // (2, 16)
    dim3 gh(kH / 64,  kM / 128);   // (48, 16)
    dim3 gv(kV / 64,  kM / 128);   // (500, 16)

    for (int i = 0; i < kDepth; ++i) {
        layernorm_kernel<<<kM, 256, 0, stream>>>(x, ln1_g + i * kE, ln1_b + i * kE, xn, xn16);

        // z = xn @ z_w^T                           (f32 out: scan reads z)
        gemm_wmma<EPI_NONE><<<gz, 256, 0, stream>>>(
            xn16, kE, zw16 + (size_t)i * kE * kE, kE, zb, nullptr, kE, kE,
            nullptr, nullptr, 0);
        // params = xn @ p_w^T   (padded to 128)    (f32 out: Cp for scan; f16: dt GEMM input)
        gemm_wmma<EPI_NONE><<<gp, 256, 0, stream>>>(
            xn16, kE, pw16 + (size_t)i * kPp * kE, kE, pb, pb16, kPp, kE,
            nullptr, nullptr, 0);
        // dt = softplus(params[:, :48] @ dtp_w^T + b)  -- K padded to 64 with zero weights
        gemm_wmma<EPI_SOFTPLUS><<<gz, 256, 0, stream>>>(
            pb16, kPp, dtw16 + (size_t)i * kE * kRp, kRp, dtb, nullptr, kE, kRp,
            dtp_b + i * kE, nullptr, 0);

        conv_silu_kernel<<<cdiv(tot, 256), 256, 0, stream>>>(
            xn, conv_w + (size_t)i * kE * kK, ub);

        dim3 gs(kE / 256, kB);
        ssm_scan_kernel<<<gs, 256, 0, stream>>>(
            pb, dtb, ub, zb, A_log + (size_t)i * kE * kN, D_param + i * kE, yg16);

        // x += (y*silu(z)) @ out_w^T
        gemm_wmma<EPI_RES><<<gz, 256, 0, stream>>>(
            yg16, kE, ow16 + (size_t)i * kE * kE, kE, x, nullptr, kE, kE,
            nullptr, x, kE);

        layernorm_kernel<<<kM, 256, 0, stream>>>(x, ln2_g + i * kE, ln2_b + i * kE, xn, xn16);

        // h = gelu(xn @ w1^T + b1)   (f16 only; feeds mlp2)
        gemm_wmma<EPI_GELU><<<gh, 256, 0, stream>>>(
            xn16, kE, w116 + (size_t)i * kH * kE, kE, nullptr, hbuf16, kH, kE,
            mlp_b1 + i * kH, nullptr, 0);
        // x += h @ w2^T + b2
        gemm_wmma<EPI_BIAS_RES><<<gz, 256, 0, stream>>>(
            hbuf16, kH, w216 + (size_t)i * kE * kH, kH, x, nullptr, kE, kH,
            mlp_b2 + i * kE, x, kE);
    }

    layernorm_kernel<<<kM, 256, 0, stream>>>(x, lnout_g, lnout_b, xn, xn16);
    gemm_wmma<EPI_BIAS><<<gv, 256, 0, stream>>>(
        xn16, kE, hw16, kE, (float*)d_out, nullptr, kV, kE,
        head_b, nullptr, 0);
}
